// ImageJSCCEncoder_21784074125565
// MI455X (gfx1250) — compile-verified, matrix operand order checked by reading, not run on hardware
//
#include <hip/hip_runtime.h>

typedef _Float16 f16;
typedef __attribute__((ext_vector_type(16))) _Float16 v16h;
typedef __attribute__((ext_vector_type(8)))  float    v8f;
typedef __attribute__((ext_vector_type(4)))  int      v4i;

// ---------------- constants ----------------
#define Bz   32
#define Hh   64
#define Ww   64
#define Cc   256
#define NH   8
#define WS   8
#define SS   4
#define Nn   64            // tokens per window
#define HD   32            // head dim
#define NWIN 2048          // B * (H/WS)*(W/WS)
#define NTOK 131072        // B*H*W
#define MLPH 1024

static __device__ __forceinline__ v8f wmma16(v16h a, v16h b, v8f c) {
  return __builtin_amdgcn_wmma_f32_16x16x32_f16(false, a, false, b, (short)0, c, false, false);
}

// A/B fragment K index for element e of v16h (documented 16-bit layout):
// k = kh + e (e<8) ; k = kh + e + 8 (e>=8)
static __device__ __forceinline__ int kmap(int kh, int e) { return kh + e + (e >= 8 ? 8 : 0); }

// ---------------- CDNA5 async global->LDS copy (ASYNCcnt path) ----------------
typedef __attribute__((address_space(1))) v4i gv4i;
typedef __attribute__((address_space(3))) v4i lv4i;

static __device__ __forceinline__ void async_copy16(const void* g, void* l) {
#if __has_builtin(__builtin_amdgcn_global_load_async_to_lds_b128)
  __builtin_amdgcn_global_load_async_to_lds_b128((gv4i*)g, (lv4i*)l, 0, 0);
#else
  *(int4*)l = *(const int4*)g;
#endif
}

static __device__ __forceinline__ void wait_async() {
#if __has_builtin(__builtin_amdgcn_global_load_async_to_lds_b128)
#if __has_builtin(__builtin_amdgcn_s_wait_asynccnt)
  __builtin_amdgcn_s_wait_asynccnt(0);
#else
  asm volatile("s_wait_asynccnt 0" ::: "memory");
#endif
#endif
}

// ---------------- fp32 -> f16 convert ----------------
__global__ void cvt_kernel(const float* __restrict__ s, f16* __restrict__ d, int n) {
  int i = blockIdx.x * 256 + threadIdx.x;
  if (i < n) d[i] = (f16)s[i];
}

// ---------------- LN1 + cyclic shift + window partition ----------------
__global__ __launch_bounds__(256) void ln1_win_kernel(const float* __restrict__ x,
                                                      const float* __restrict__ gw,
                                                      const float* __restrict__ gb,
                                                      f16* __restrict__ xw) {
  int g = blockIdx.x * 8 + (threadIdx.x >> 5);
  int lane = threadIdx.x & 31;
  int w = g >> 6, t = g & 63;
  int b = w >> 6, wi = w & 63;
  int rs = (wi >> 3) * WS + (t >> 3);
  int cs = (wi & 7) * WS + (t & 7);
  int r = (rs + SS) & 63, c = (cs + SS) & 63;
  const float* xp = x + ((size_t)b * 4096 + r * 64 + c) * Cc;
  float v[8];
  float mu = 0.f;
#pragma unroll
  for (int i = 0; i < 8; ++i) { v[i] = xp[lane + i * 32]; mu += v[i]; }
#pragma unroll
  for (int s = 1; s < 32; s <<= 1) mu += __shfl_xor(mu, s, 32);
  mu *= (1.f / 256.f);
  float var = 0.f;
#pragma unroll
  for (int i = 0; i < 8; ++i) { float d = v[i] - mu; var += d * d; }
#pragma unroll
  for (int s = 1; s < 32; s <<= 1) var += __shfl_xor(var, s, 32);
  var *= (1.f / 256.f);
  float inv = rsqrtf(var + 1e-5f);
#pragma unroll
  for (int i = 0; i < 8; ++i) {
    int e = lane + i * 32;
    xw[(size_t)g * Cc + e] = (f16)((v[i] - mu) * inv * gw[e] + gb[e]);
  }
}

// ---------------- LN2 (token order) ----------------
__global__ __launch_bounds__(256) void ln2_kernel(const float* __restrict__ x1,
                                                  const float* __restrict__ gw,
                                                  const float* __restrict__ gb,
                                                  f16* __restrict__ h2) {
  int g = blockIdx.x * 8 + (threadIdx.x >> 5);
  int lane = threadIdx.x & 31;
  const float* xp = x1 + (size_t)g * Cc;
  float v[8];
  float mu = 0.f;
#pragma unroll
  for (int i = 0; i < 8; ++i) { v[i] = xp[lane + i * 32]; mu += v[i]; }
#pragma unroll
  for (int s = 1; s < 32; s <<= 1) mu += __shfl_xor(mu, s, 32);
  mu *= (1.f / 256.f);
  float var = 0.f;
#pragma unroll
  for (int i = 0; i < 8; ++i) { float d = v[i] - mu; var += d * d; }
#pragma unroll
  for (int s = 1; s < 32; s <<= 1) var += __shfl_xor(var, s, 32);
  var *= (1.f / 256.f);
  float inv = rsqrtf(var + 1e-5f);
#pragma unroll
  for (int i = 0; i < 8; ++i) {
    int e = lane + i * 32;
    h2[(size_t)g * Cc + e] = (f16)((v[i] - mu) * inv * gw[e] + gb[e]);
  }
}

// ---------------- generic WMMA GEMM: Y[M,N] = A[M,K] * Wt[N,K]^T (+epilogue) ----
// mode 0: outh = f16(acc + bias)                 (QKV)
// mode 1: window-reverse scatter + residual      (proj):  x1 = proj + x
// mode 2: outh = f16(gelu(acc + bias))           (FC1)
// mode 3: outf = acc + bias + res                (FC2 -> final output)
#define BM 128
#define BN 128
#define BK 32
__global__ __launch_bounds__(256) void gemm_kernel(const f16* __restrict__ A,
                                                   const f16* __restrict__ Wt,
                                                   const float* __restrict__ bias,
                                                   int M, int N, int K, int mode,
                                                   f16* __restrict__ outh, int ldo,
                                                   float* __restrict__ outf,
                                                   const float* __restrict__ res) {
  __shared__ f16 As[BM][BK];
  __shared__ f16 Bs[BN][BK];
  int tid = threadIdx.x;
  int wave = tid >> 5, lane = tid & 31;
  int wm = wave >> 2, wn = wave & 3;            // 2x4 wave grid -> 64x32 per wave
  int bm0 = blockIdx.x * BM, bn0 = blockIdx.y * BN;
  int nl = lane & 15, kh = (lane >> 4) * 8, mh = lane >> 4;
  v8f acc[4][2] = {};

  for (int k0 = 0; k0 < K; k0 += BK) {
    // async DMA the 128x32 A tile and 128x32 Wt tile straight into LDS
#pragma unroll
    for (int it = 0; it < 2; ++it) {
      int idx = tid + it * 256;                 // 0..511 -> 128 rows x 4 chunks
      int r = idx >> 2, cc = (idx & 3) * 8;
      async_copy16(&A[(size_t)(bm0 + r) * K + k0 + cc], &As[r][cc]);
      async_copy16(&Wt[(size_t)(bn0 + r) * K + k0 + cc], &Bs[r][cc]);
    }
    if (k0 + BK < K) {                          // next-tile prefetch (global_prefetch_b8)
      __builtin_prefetch(&A[(size_t)(bm0 + (tid >> 1)) * K + k0 + BK], 0, 1);
      __builtin_prefetch(&Wt[(size_t)(bn0 + (tid >> 1)) * K + k0 + BK], 0, 1);
    }
    wait_async();
    __syncthreads();

    v16h bf[2];
#pragma unroll
    for (int nj = 0; nj < 2; ++nj) {
      int n = wn * 32 + nj * 16 + nl;
      v16h t;
#pragma unroll
      for (int e = 0; e < 16; ++e) t[e] = Bs[n][kmap(kh, e)];
      bf[nj] = t;
    }
#pragma unroll
    for (int mi = 0; mi < 4; ++mi) {
      int m = wm * 64 + mi * 16 + nl;
      v16h a;
#pragma unroll
      for (int e = 0; e < 16; ++e) a[e] = As[m][kmap(kh, e)];
      acc[mi][0] = wmma16(a, bf[0], acc[mi][0]);
      acc[mi][1] = wmma16(a, bf[1], acc[mi][1]);
    }
    __syncthreads();
  }

  // epilogue: D layout -> lane holds column nl; rows r + 8*mh
#pragma unroll
  for (int mi = 0; mi < 4; ++mi) {
#pragma unroll
    for (int nj = 0; nj < 2; ++nj) {
      int gn = bn0 + wn * 32 + nj * 16 + nl;
      float bb = bias ? bias[gn] : 0.f;
#pragma unroll
      for (int r = 0; r < 8; ++r) {
        int gm = bm0 + wm * 64 + mi * 16 + mh * 8 + r;
        float vv = acc[mi][nj][r] + bb;
        if (mode == 0) {
          outh[(size_t)gm * ldo + gn] = (f16)vv;
        } else if (mode == 1) {
          int w = gm >> 6, t = gm & 63;
          int b = w >> 6, wi = w & 63;
          int rs = (wi >> 3) * WS + (t >> 3);
          int cs = (wi & 7) * WS + (t & 7);
          int rr = (rs + SS) & 63, ccp = (cs + SS) & 63;
          size_t di = ((size_t)b * 4096 + rr * 64 + ccp) * Cc + gn;
          outf[di] = vv + res[di];
        } else if (mode == 2) {
          float gl = 0.5f * vv * (1.0f + erff(vv * 0.70710678118654752f));
          outh[(size_t)gm * ldo + gn] = (f16)gl;
        } else {
          size_t di = (size_t)gm * Cc + gn;
          outf[di] = vv + res[di];
        }
      }
    }
  }
}

// ---------------- windowed attention: 1 block per window, 1 wave per head ----
static __device__ __forceinline__ int regid(int wblk, int i) {
  int p = wblk * WS + i;              // pixel coord in shifted image
  return p < (Hh - WS) ? 0 : (p < (Hh - SS) ? 1 : 2);
}

__global__ __launch_bounds__(256) void attn_kernel(const f16* __restrict__ qkv,
                                                   const float* __restrict__ rpb,
                                                   f16* __restrict__ outw) {
  __shared__ f16   P[NH][16][72];      // per-head prob tile (16 rows x 64 keys, padded)
  __shared__ float rpbs[225 * NH];     // staged relative-position-bias table (7.2 KB)
  for (int i = threadIdx.x; i < 225 * NH; i += 256) rpbs[i] = rpb[i];
  __syncthreads();

  int w = blockIdx.x;                  // window 0..2047
  int wi = w & 63;
  int head = threadIdx.x >> 5, lane = threadIdx.x & 31;
  int nl = lane & 15, kh = (lane >> 4) * 8, mhh = lane >> 4;
  const f16* base = qkv + (size_t)w * Nn * (3 * Cc);
  const int ld = 3 * Cc;

  // K fragments (B layout): tile nj -> B[k=hd][n=key]
  v16h kf[4];
#pragma unroll
  for (int nj = 0; nj < 4; ++nj) {
    int key = nj * 16 + nl;
    const f16* kp = base + (size_t)key * ld + Cc + head * HD;
    v16h t;
#pragma unroll
    for (int e = 0; e < 16; ++e) t[e] = kp[kmap(kh, e)];
    kf[nj] = t;
  }
  // V fragments (B layout): ks = K chunk over tokens, nj over head-dim cols
  v16h vf[2][2];
#pragma unroll
  for (int ks = 0; ks < 2; ++ks)
#pragma unroll
    for (int nj = 0; nj < 2; ++nj) {
      int hd = nj * 16 + nl;
      v16h t;
#pragma unroll
      for (int e = 0; e < 16; ++e) {
        int kk = ks * 32 + kmap(kh, e);
        t[e] = base[(size_t)kk * ld + 2 * Cc + head * HD + hd];
      }
      vf[ks][nj] = t;
    }

  const float scale = 0.17677669529663687f;   // 1/sqrt(32)
  int wh = wi >> 3, wwc = wi & 7;

  for (int mi = 0; mi < 4; ++mi) {
    // Q A-fragment for rows mi*16..mi*16+15, scale folded in
    v16h qa;
    {
      int m = mi * 16 + nl;
      const f16* qp = base + (size_t)m * ld + head * HD;
#pragma unroll
      for (int e = 0; e < 16; ++e) qa[e] = (f16)((float)qp[kmap(kh, e)] * scale);
    }
    v8f at[4];
#pragma unroll
    for (int nj = 0; nj < 4; ++nj) { v8f z = {}; at[nj] = wmma16(qa, kf[nj], z); }

    // bias + shift-mask + softmax (row = 64 wide, lives in 16 lanes x 4 tiles)
#pragma unroll
    for (int r = 0; r < 8; ++r) {
      int m = mi * 16 + mhh * 8 + r;
      int qi = m >> 3, qj = m & 7;
      int rq = regid(wh, qi) * 3 + regid(wwc, qj);
#pragma unroll
      for (int nj = 0; nj < 4; ++nj) {
        int n = nj * 16 + nl;
        int ki = n >> 3, kj = n & 7;
        int rk = regid(wh, ki) * 3 + regid(wwc, kj);
        int ridx = (qi - ki + 7) * 15 + (qj - kj + 7);
        at[nj][r] += rpbs[ridx * NH + head] + ((rq != rk) ? -100.0f : 0.0f);
      }
      float mx = fmaxf(fmaxf(at[0][r], at[1][r]), fmaxf(at[2][r], at[3][r]));
#pragma unroll
      for (int s = 1; s < 16; s <<= 1) mx = fmaxf(mx, __shfl_xor(mx, s, 32));
      float sum = 0.f;
#pragma unroll
      for (int nj = 0; nj < 4; ++nj) { float e = __expf(at[nj][r] - mx); at[nj][r] = e; sum += e; }
#pragma unroll
      for (int s = 1; s < 16; s <<= 1) sum += __shfl_xor(sum, s, 32);
      float inv = 1.0f / sum;
#pragma unroll
      for (int nj = 0; nj < 4; ++nj)
        P[head][mhh * 8 + r][nj * 16 + nl] = (f16)(at[nj][r] * inv);
    }
    asm volatile("s_wait_dscnt 0" ::: "memory");   // wave-local LDS RAW fence

    // P -> A fragments, then PV
    v16h pa[2];
#pragma unroll
    for (int ks = 0; ks < 2; ++ks) {
      v16h t;
#pragma unroll
      for (int e = 0; e < 16; ++e) t[e] = P[head][nl][ks * 32 + kmap(kh, e)];
      pa[ks] = t;
    }
    v8f o0 = {}, o1 = {};
    o0 = wmma16(pa[0], vf[0][0], o0);
    o0 = wmma16(pa[1], vf[1][0], o0);
    o1 = wmma16(pa[0], vf[0][1], o1);
    o1 = wmma16(pa[1], vf[1][1], o1);
#pragma unroll
    for (int r = 0; r < 8; ++r) {
      int m = mi * 16 + mhh * 8 + r;
      size_t ro = ((size_t)w * Nn + m) * Cc + head * HD;
      outw[ro + nl]      = (f16)o0[r];
      outw[ro + 16 + nl] = (f16)o1[r];
    }
  }
}

// ---------------- workspace layout (bytes) ----------------
// region A (reused):  [0, 268435456)
//   xw      f16 [131072,256]  @ 0            (dead after QKV)
//   qkv     f16 [131072,768]  @ 67108864     (dead after attention)
//   attnout f16 [131072,256]  @ 0            (reuses xw; dead after proj)
//   hid     f16 [131072,1024] @ 0            (reuses whole region A)
// x1  f32 [131072,256] @ 268435456
// h2  f16 [131072,256] @ 402653184
// f16 weights          @ 469762048

extern "C" void kernel_launch(void* const* d_in, const int* in_sizes, int n_in,
                              void* d_out, int out_size, void* d_ws, size_t ws_size,
                              hipStream_t stream) {
  (void)in_sizes; (void)n_in; (void)out_size; (void)ws_size;
  const float* x       = (const float*)d_in[0];
  const float* norm1_w = (const float*)d_in[1];
  const float* norm1_b = (const float*)d_in[2];
  const float* qkv_w   = (const float*)d_in[3];
  const float* qkv_b   = (const float*)d_in[4];
  const float* rpb     = (const float*)d_in[5];
  const float* proj_w  = (const float*)d_in[6];
  const float* proj_b  = (const float*)d_in[7];
  const float* norm2_w = (const float*)d_in[8];
  const float* norm2_b = (const float*)d_in[9];
  const float* fc1_w   = (const float*)d_in[10];
  const float* fc1_b   = (const float*)d_in[11];
  const float* fc2_w   = (const float*)d_in[12];
  const float* fc2_b   = (const float*)d_in[13];
  float* out = (float*)d_out;

  char* ws = (char*)d_ws;
  f16*   xw   = (f16*)(ws + 0);
  f16*   qkv  = (f16*)(ws + 67108864ull);
  f16*   aout = (f16*)(ws + 0);
  f16*   hid  = (f16*)(ws + 0);
  float* x1   = (float*)(ws + 268435456ull);
  f16*   h2   = (f16*)(ws + 402653184ull);
  f16*   qkvw16  = (f16*)(ws + 469762048ull);
  f16*   projw16 = (f16*)(ws + 469762048ull + 393216ull);
  f16*   fc1w16  = (f16*)(ws + 469762048ull + 524288ull);
  f16*   fc2w16  = (f16*)(ws + 469762048ull + 1048576ull);

  // weight converts
  cvt_kernel<<<(768 * 256 + 255) / 256, 256, 0, stream>>>(qkv_w, qkvw16, 768 * 256);
  cvt_kernel<<<(256 * 256 + 255) / 256, 256, 0, stream>>>(proj_w, projw16, 256 * 256);
  cvt_kernel<<<(1024 * 256 + 255) / 256, 256, 0, stream>>>(fc1_w, fc1w16, 1024 * 256);
  cvt_kernel<<<(256 * 1024 + 255) / 256, 256, 0, stream>>>(fc2_w, fc2w16, 256 * 1024);

  // LN1 + shift + window partition
  ln1_win_kernel<<<NTOK / 8, 256, 0, stream>>>(x, norm1_w, norm1_b, xw);

  // QKV: [131072,256] x [768,256]^T
  gemm_kernel<<<dim3(NTOK / BM, 768 / BN), 256, 0, stream>>>(
      xw, qkvw16, qkv_b, NTOK, 768, 256, 0, qkv, 768, nullptr, nullptr);

  // windowed attention
  attn_kernel<<<NWIN, 256, 0, stream>>>(qkv, rpb, aout);

  // proj + window-reverse/unshift + residual -> x1
  gemm_kernel<<<dim3(NTOK / BM, 256 / BN), 256, 0, stream>>>(
      aout, projw16, proj_b, NTOK, 256, 256, 1, nullptr, 256, x1, x);

  // LN2
  ln2_kernel<<<NTOK / 8, 256, 0, stream>>>(x1, norm2_w, norm2_b, h2);

  // FC1 + GELU
  gemm_kernel<<<dim3(NTOK / BM, MLPH / BN), 256, 0, stream>>>(
      h2, fc1w16, fc1_b, NTOK, MLPH, 256, 2, hid, MLPH, nullptr, nullptr);

  // FC2 + residual -> output
  gemm_kernel<<<dim3(NTOK / BM, 256 / BN), 256, 0, stream>>>(
      hid, fc2w16, fc2_b, NTOK, 256, MLPH, 3, nullptr, 256, out, x1);
}
